// StackGCN_56203942036103
// MI455X (gfx1250) — matrix operand order repaired and checked by reading, not compiled, not run on hardware
//
#include <hip/hip_runtime.h>
#include <hip/hip_bf16.h>

// ---------------- problem constants (match reference) ----------------
constexpr int NUM_USERS   = 50000;
constexpr int NUM_ITEMS   = 10000;
constexpr int NUM_SUPPORT = 5;
constexpr int INPUT_DIM   = 512;   // K
constexpr int OUTPUT_DIM  = 500;   // N (padded to 512 internally)
constexpr int NPAD        = 512;
constexpr int NNZ         = 400000;
constexpr int HIDDEN      = 100;

// GEMM tiling
constexpr int BM = 128, BN = 128, BK = 32;
constexpr int LDST = 40;                       // LDS row stride (bf16): 80B -> conflict-free b128
constexpr int MPAD_U = ((NUM_USERS + BM - 1) / BM) * BM;   // 50048
constexpr int MPAD_V = ((NUM_ITEMS + BM - 1) / BM) * BM;   // 10112

typedef __attribute__((ext_vector_type(16))) __bf16 v16bf;
typedef __attribute__((ext_vector_type(8)))  float  v8f;

union Frag  { v16bf f; uint4 u[2]; };
union BF8   { __bf16 h[8]; uint4 u; };

// ---------------- CDNA5 async global->LDS copy (ASYNCcnt path) ----------------
__device__ __forceinline__ void async_copy_b128(unsigned lds_off, const void* gaddr) {
    // lds_off gives the LDS byte address; copies 16B per lane, tracked by ASYNCcnt
    asm volatile("global_load_async_to_lds_b128 %0, %1, off"
                 :: "v"(lds_off), "v"(gaddr) : "memory");
}
__device__ __forceinline__ void wait_async0() {
    asm volatile("s_wait_asynccnt 0x0" ::: "memory");
}
// low 32 bits of a generic pointer to __shared__ == LDS byte offset
__device__ __forceinline__ unsigned lds_offset(const void* p) {
    return (unsigned)(unsigned long long)p;
}

// ---------------- utility kernels ----------------
__global__ void zero_f32v4(float4* __restrict__ p, long long n4) {
    long long i = (long long)blockIdx.x * blockDim.x + threadIdx.x;
    long long s = (long long)gridDim.x * blockDim.x;
    float4 z = make_float4(0.f, 0.f, 0.f, 0.f);
    for (; i < n4; i += s) p[i] = z;
}

__global__ void relu_f32v4(float4* __restrict__ p, long long n4) {
    long long i = (long long)blockIdx.x * blockDim.x + threadIdx.x;
    long long s = (long long)gridDim.x * blockDim.x;
    for (; i < n4; i += s) {
        float4 v = p[i];
        v.x = fmaxf(v.x, 0.f); v.y = fmaxf(v.y, 0.f);
        v.z = fmaxf(v.z, 0.f); v.w = fmaxf(v.w, 0.f);
        p[i] = v;
    }
}

// W [K=512, N=500] f32 row-major -> Wt bf16 [NPAD][K] (transposed, zero-padded)
__global__ void convert_w(const float* __restrict__ W, __bf16* __restrict__ Wt) {
    int idx = blockIdx.x * blockDim.x + threadIdx.x;  // n*K + k
    if (idx >= NPAD * INPUT_DIM) return;
    int n = idx >> 9;
    int k = idx & (INPUT_DIM - 1);
    float v = (n < OUTPUT_DIM) ? W[k * OUTPUT_DIM + n] : 0.0f;
    Wt[idx] = (__bf16)v;
}

// X [M,512] f32 -> Xb bf16 [Mpad,512], rows >= M zero-filled. 8 elems/thread.
__global__ void convert_x(const float* __restrict__ X, __bf16* __restrict__ Xb,
                          int M, int Mpad) {
    long long t = (long long)blockIdx.x * blockDim.x + threadIdx.x;
    long long total = (long long)Mpad * INPUT_DIM / 8;
    if (t >= total) return;
    long long e = t * 8;
    int row = (int)(e >> 9);
    int k   = (int)(e & (INPUT_DIM - 1));
    BF8 o;
    if (row < M) {
        float4 a = *(const float4*)(X + (size_t)row * INPUT_DIM + k);
        float4 b = *(const float4*)(X + (size_t)row * INPUT_DIM + k + 4);
        o.h[0] = (__bf16)a.x; o.h[1] = (__bf16)a.y;
        o.h[2] = (__bf16)a.z; o.h[3] = (__bf16)a.w;
        o.h[4] = (__bf16)b.x; o.h[5] = (__bf16)b.y;
        o.h[6] = (__bf16)b.z; o.h[7] = (__bf16)b.w;
    } else {
        o.u = make_uint4(0u, 0u, 0u, 0u);
    }
    *(uint4*)(Xb + e) = o.u;
}

// ---------------- bf16 WMMA GEMM: OUT[Mpad,NPAD] = Xb * Wt^T ----------------
// 256 threads = 8 waves (4 M x 2 N); wave tile 32x64 = 8 accumulators.
// Double-buffered LDS filled by global_load_async_to_lds_b128.
__global__ __launch_bounds__(256) void gemm_xw_bf16(
    const __bf16* __restrict__ Xb, const __bf16* __restrict__ Wt,
    float* __restrict__ OUT) {
    __shared__ __align__(16) __bf16 As[2][BM * LDST];
    __shared__ __align__(16) __bf16 Bs[2][BN * LDST];

    const int tid  = threadIdx.x;
    const int lane = tid & 31;
    const int wave = tid >> 5;
    const int m0 = blockIdx.x * BM;
    const int n0 = blockIdx.y * BN;
    const int wm = (wave >> 1) * 32;   // 0,32,64,96
    const int wn = (wave & 1) * 64;    // 0,64

    const int fm = lane & 15;          // row/col inside 16-wide tile
    const int fh = lane >> 4;          // lane-half K select

    // staging: 512 chunks of 16B per matrix; thread handles chunks tid, tid+256
    const int r0 = tid >> 2;                 // 0..63
    const int c0 = (tid & 3) * 8;            // bf16 elems: 0,8,16,24
    const unsigned aoff0 = lds_offset(&As[0][0]);
    const unsigned boff0 = lds_offset(&Bs[0][0]);
    const unsigned bufstride = (unsigned)(BM * LDST * sizeof(__bf16));

    auto issue_tile = [&](int k0, int buf) {
        const __bf16* ga = Xb + (size_t)(m0 + r0) * INPUT_DIM + k0 + c0;
        const __bf16* gb = Wt + (size_t)(n0 + r0) * INPUT_DIM + k0 + c0;
        unsigned la = aoff0 + buf * bufstride + (unsigned)(r0 * LDST + c0) * 2u;
        unsigned lb = boff0 + buf * bufstride + (unsigned)(r0 * LDST + c0) * 2u;
        async_copy_b128(la, ga);
        async_copy_b128(lb, gb);
        // second half: rows r0+64
        const __bf16* ga2 = ga + (size_t)64 * INPUT_DIM;
        const __bf16* gb2 = gb + (size_t)64 * INPUT_DIM;
        async_copy_b128(la + 64u * LDST * 2u, ga2);
        async_copy_b128(lb + 64u * LDST * 2u, gb2);
    };

    v8f acc[2][4] = {};

    issue_tile(0, 0);

    for (int k0 = 0; k0 < INPUT_DIM; k0 += BK) {
        const int buf = (k0 / BK) & 1;
        wait_async0();          // this wave's copies into `buf` landed
        __syncthreads();        // everyone's copies landed; prev reads retired
        if (k0 + BK < INPUT_DIM) issue_tile(k0 + BK, buf ^ 1);

        // fragment loads (ds_read_b128 x2 each) + 8 WMMAs
        Frag a[2], b[4];
        for (int i = 0; i < 2; ++i) {
            const __bf16* pa = &As[buf][(wm + i * 16 + fm) * LDST + fh * 8];
            a[i].u[0] = *(const uint4*)pa;
            a[i].u[1] = *(const uint4*)(pa + 16);
        }
        for (int j = 0; j < 4; ++j) {
            const __bf16* pb = &Bs[buf][(wn + j * 16 + fm) * LDST + fh * 8];
            b[j].u[0] = *(const uint4*)pb;
            b[j].u[1] = *(const uint4*)(pb + 16);
        }
        for (int i = 0; i < 2; ++i)
            for (int j = 0; j < 4; ++j)
                acc[i][j] = __builtin_amdgcn_wmma_f32_16x16x32_bf16(
                    false, a[i].f, false, b[j].f, (short)0, acc[i][j],
                    false, false);
        __syncthreads();        // all reads of `buf` done before it is refilled
    }

    // store C: VGPR r -> M = r + 8*fh, N = fm  (OUT is padded; no guards)
    for (int i = 0; i < 2; ++i)
        for (int j = 0; j < 4; ++j) {
            int col = n0 + wn + j * 16 + fm;
            size_t rowbase = (size_t)(m0 + wm + i * 16 + fh * 8);
            for (int r = 0; r < 8; ++r)
                OUT[(rowbase + r) * NPAD + col] = acc[i][j][r];
        }
}

// ---------------- edge scatter: one wave per nnz ----------------
__global__ __launch_bounds__(256) void scatter_edges(
    const float* __restrict__ vals, const int* __restrict__ rows,
    const int* __restrict__ cols, const float* __restrict__ tmp_u,
    const float* __restrict__ tmp_v, float* __restrict__ z_u,
    float* __restrict__ z_v) {
    long long wid = ((long long)blockIdx.x * blockDim.x + threadIdx.x) >> 5;
    int lane = threadIdx.x & 31;
    if (wid >= (long long)NUM_SUPPORT * NNZ) return;
    int sup = (int)(wid / NNZ);
    int idx = (int)wid;

    float v = vals[idx];
    int r = rows[idx];
    int c = cols[idx];
    int hb = sup * HIDDEN;

    const float* tv = tmp_v + (size_t)c * NPAD + hb;
    const float* tu = tmp_u + (size_t)r * NPAD + hb;
    float* zu = z_u + (size_t)r * OUTPUT_DIM + hb;
    float* zv = z_v + (size_t)c * OUTPUT_DIM + hb;

    for (int h = lane; h < HIDDEN; h += 32) {
        atomicAdd(zu + h, v * tv[h]);   // global_atomic_add_f32, L2-resident
        atomicAdd(zv + h, v * tu[h]);
    }
}

// ---------------- launcher ----------------
extern "C" void kernel_launch(void* const* d_in, const int* in_sizes, int n_in,
                              void* d_out, int out_size, void* d_ws, size_t ws_size,
                              hipStream_t stream) {
    const float* x_u      = (const float*)d_in[0];
    const float* x_v      = (const float*)d_in[1];
    const float* W        = (const float*)d_in[2];
    const float* sup_vals = (const float*)d_in[3];
    const int*   sup_rows = (const int*)d_in[4];
    const int*   sup_cols = (const int*)d_in[5];

    float* z_u = (float*)d_out;
    float* z_v = z_u + (size_t)NUM_USERS * OUTPUT_DIM;

    // workspace layout (16B-aligned):
    //   Wt   bf16 [512][512]            0.5 MB
    //   Xu_b bf16 [MPAD_U][512]        51.2 MB
    //   Xv_b bf16 [MPAD_V][512]        10.4 MB
    //   tmp_u f32 [MPAD_U][512]       102.5 MB
    //   tmp_v f32 [MPAD_V][512]        20.7 MB   (total ~177 MiB)
    char* ws = (char*)d_ws;
    __bf16* Wt   = (__bf16*)ws;
    __bf16* Xu_b = (__bf16*)(ws + (size_t)NPAD * INPUT_DIM * 2);
    __bf16* Xv_b = Xu_b + (size_t)MPAD_U * INPUT_DIM;
    float*  tmp_u = (float*)(Xv_b + (size_t)MPAD_V * INPUT_DIM);
    float*  tmp_v = tmp_u + (size_t)MPAD_U * NPAD;

    const long long outN =
        (long long)NUM_USERS * OUTPUT_DIM + (long long)NUM_ITEMS * OUTPUT_DIM;

    // 1) zero accumulation target
    zero_f32v4<<<4096, 256, 0, stream>>>((float4*)z_u, outN / 4);

    // 2) precision conversion (one cheap bandwidth pass)
    convert_w<<<(NPAD * INPUT_DIM + 255) / 256, 256, 0, stream>>>(W, Wt);
    convert_x<<<((long long)MPAD_U * INPUT_DIM / 8 + 255) / 256, 256, 0, stream>>>(
        x_u, Xu_b, NUM_USERS, MPAD_U);
    convert_x<<<((long long)MPAD_V * INPUT_DIM / 8 + 255) / 256, 256, 0, stream>>>(
        x_v, Xv_b, NUM_ITEMS, MPAD_V);

    // 3) dense projections via bf16 WMMA (async-LDS double-buffered)
    dim3 gu(MPAD_U / BM, NPAD / BN);
    gemm_xw_bf16<<<gu, 256, 0, stream>>>(Xu_b, Wt, tmp_u);
    dim3 gv(MPAD_V / BM, NPAD / BN);
    gemm_xw_bf16<<<gv, 256, 0, stream>>>(Xv_b, Wt, tmp_v);

    // 4) sparse scatter (one wave per edge)
    long long waves  = (long long)NUM_SUPPORT * NNZ;
    int       blocks = (int)((waves * 32 + 255) / 256);
    scatter_edges<<<blocks, 256, 0, stream>>>(sup_vals, sup_rows, sup_cols,
                                              tmp_u, tmp_v, z_u, z_v);

    // 5) ReLU in place
    relu_f32v4<<<4096, 256, 0, stream>>>((float4*)z_u, outN / 4);
}
